// GRU_66743791780056
// MI455X (gfx1250) — compile-verified, matrix operand order checked by reading, not run on hardware
//
#include <hip/hip_runtime.h>

// ---------------------------------------------------------------------------
// GRU on MI455X (gfx1250): bf16 WMMA (v_wmma_f32_16x16x32_bf16), f32 accum.
//   Phase 0: convert weights/x/h0 -> bf16 (packed for WMMA B-fragments).
//   Phase 1: precompute Gg[t,b,:] = x_t @ Wg_x.T + b_g, g in {z,r,t}
//            (time-parallel GEMM, 4 N-tiles per wave, pipelined loads).
//   Phase 2: sequential scan, 2 kernels per step (software-pipelined K loop,
//            pure-bf16 fragments on the serial path):
//            A: zr_pre = h_prev @ Wzr_h.T (+G) -> z, r, rh = r*h_prev
//            B: h~ = tanh(rh @ Wt_h.T + Gt); h = (1-z)h + z h~  (+ write hbf)
// ---------------------------------------------------------------------------

typedef __bf16 bf16_t;
typedef bf16_t v16bf __attribute__((ext_vector_type(16)));
typedef bf16_t v8bf  __attribute__((ext_vector_type(8)));
typedef bf16_t v4bf  __attribute__((ext_vector_type(4)));
typedef float  v8f   __attribute__((ext_vector_type(8)));

#define GRU_T  512
#define GRU_B  64
#define GRU_H  1024
#define GRU_I  1024
#define GRU_HI (GRU_H + GRU_I)

#define WMMA_BF16(a, b, c) \
  __builtin_amdgcn_wmma_f32_16x16x32_bf16(false, (a), false, (b), (short)0, (c), false, false)

// A-fragment (16x32, MxK) / B-fragment (32x16 col-major == row n of row-major W):
// lane L: row/col = L&15, K-halves at (L>>4)*8 and (L>>4)*8+16. Two 16B loads.
static __device__ __forceinline__ v16bf frag_bf16(const bf16_t* __restrict__ p,
                                                  int stride, int lane) {
  const int idx = lane & 15;
  const int kh  = (lane >> 4) << 3;
  const bf16_t* r = p + (size_t)idx * stride + kh;
  const v8bf lo = *(const v8bf*)(r);
  const v8bf hi = *(const v8bf*)(r + 16);
  v16bf f;
#pragma unroll
  for (int i = 0; i < 8; ++i) { f[i] = lo[i]; f[8 + i] = hi[i]; }
  return f;
}

static __device__ __forceinline__ float fast_sigmoid(float x) {
  return 1.0f / (1.0f + __expf(-x));
}

// ---------------------------------------------------------------------------
// Phase 0a: weight conversion / packing (6M elements).
// ---------------------------------------------------------------------------
__global__ void gru_convert_weights(const float* __restrict__ Wz,
                                    const float* __restrict__ Wr,
                                    const float* __restrict__ Wt,
                                    bf16_t* __restrict__ Wzr_h,
                                    bf16_t* __restrict__ Wt_h,
                                    bf16_t* __restrict__ Wzx,
                                    bf16_t* __restrict__ Wrx,
                                    bf16_t* __restrict__ Wtx) {
  const int tid = blockIdx.x * blockDim.x + threadIdx.x;   // [0, 6*2^20)
  const int mat = tid >> 20;
  const int n   = (tid >> 10) & 1023;
  const int k   = tid & 1023;
  float v;
  bf16_t* dst;
  switch (mat) {
    case 0:  v = Wz[(size_t)n * GRU_HI + k];          dst = Wzr_h + (size_t)n * GRU_H + k;            break;
    case 1:  v = Wr[(size_t)n * GRU_HI + k];          dst = Wzr_h + (size_t)(GRU_H + n) * GRU_H + k;  break;
    case 2:  v = Wt[(size_t)n * GRU_HI + k];          dst = Wt_h  + (size_t)n * GRU_H + k;            break;
    case 3:  v = Wz[(size_t)n * GRU_HI + GRU_H + k];  dst = Wzx   + (size_t)n * GRU_I + k;            break;
    case 4:  v = Wr[(size_t)n * GRU_HI + GRU_H + k];  dst = Wrx   + (size_t)n * GRU_I + k;            break;
    default: v = Wt[(size_t)n * GRU_HI + GRU_H + k];  dst = Wtx   + (size_t)n * GRU_I + k;            break;
  }
  *dst = (bf16_t)v;
}

// Phase 0b: x f32 -> bf16 (4 elements/thread; b128 in, b64 out).
__global__ void gru_convert_x(const float* __restrict__ x,
                              bf16_t* __restrict__ xbf) {
  const size_t i = ((size_t)blockIdx.x * blockDim.x + threadIdx.x) * 4;
  const float4 v = *(const float4*)(x + i);
  v4bf o;
  o[0] = (bf16_t)v.x; o[1] = (bf16_t)v.y; o[2] = (bf16_t)v.z; o[3] = (bf16_t)v.w;
  *(v4bf*)(xbf + i) = o;
}

// Phase 0c: h0 f32 -> bf16.
__global__ void gru_convert_h(const float* __restrict__ h0,
                              bf16_t* __restrict__ hbf) {
  const int i = blockIdx.x * blockDim.x + threadIdx.x;   // 64K
  hbf[i] = (bf16_t)h0[i];
}

// ---------------------------------------------------------------------------
// Phase 1: Gg = x @ Wg_x.T + b_g for all T at once. One wave per (16 x 64)
// strip: 4 N-tiles per wave, A-fragment reused, distance-1 pipelined loads.
// waves = 3 * 2048 * 16 = 98304 -> 12288 blocks x 8 waves.
// ---------------------------------------------------------------------------
__global__ void gru_gatepre(const bf16_t* __restrict__ xbf,
                            const bf16_t* __restrict__ Wzx,
                            const bf16_t* __restrict__ Wrx,
                            const bf16_t* __restrict__ Wtx,
                            const float* __restrict__ bz,
                            const float* __restrict__ br,
                            const float* __restrict__ bt,
                            bf16_t* __restrict__ Gz,
                            bf16_t* __restrict__ Gr,
                            bf16_t* __restrict__ Gt) {
  const int lane  = threadIdx.x & 31;
  const int wave  = (blockIdx.x * blockDim.x + threadIdx.x) >> 5;
  const int MT    = (GRU_T * GRU_B) / 16;   // 2048
  const int NG    = (GRU_H / 16) / 4;       // 16
  const int g     = wave / (MT * NG);
  const int rem   = wave % (MT * NG);
  const int mt    = rem / NG;
  const int ng    = rem % NG;

  const bf16_t* W = (g == 0) ? Wzx : (g == 1) ? Wrx : Wtx;
  const float*  b = (g == 0) ? bz  : (g == 1) ? br  : bt;
  bf16_t*       G = (g == 0) ? Gz  : (g == 1) ? Gr  : Gt;

  const bf16_t* Ab = xbf + (size_t)(mt * 16) * GRU_I;
  const bf16_t* B0 = W + (size_t)(ng * 64) * GRU_I;
  const bf16_t* B1 = B0 + (size_t)16 * GRU_I;
  const bf16_t* B2 = B0 + (size_t)32 * GRU_I;
  const bf16_t* B3 = B0 + (size_t)48 * GRU_I;

  v8f acc0 = {}, acc1 = {}, acc2 = {}, acc3 = {};
  // pipeline stage 0
  v16bf af = frag_bf16(Ab, GRU_I, lane);
  v16bf f0 = frag_bf16(B0, GRU_I, lane);
  v16bf f1 = frag_bf16(B1, GRU_I, lane);
  v16bf f2 = frag_bf16(B2, GRU_I, lane);
  v16bf f3 = frag_bf16(B3, GRU_I, lane);
#pragma unroll 4
  for (int kb = 0; kb < GRU_I; kb += 32) {
    const int kn = (kb + 32 < GRU_I) ? (kb + 32) : 0;   // keep loads unconditional
    const v16bf afn = frag_bf16(Ab + kn, GRU_I, lane);
    const v16bf f0n = frag_bf16(B0 + kn, GRU_I, lane);
    const v16bf f1n = frag_bf16(B1 + kn, GRU_I, lane);
    const v16bf f2n = frag_bf16(B2 + kn, GRU_I, lane);
    const v16bf f3n = frag_bf16(B3 + kn, GRU_I, lane);
    acc0 = WMMA_BF16(af, f0, acc0);
    acc1 = WMMA_BF16(af, f1, acc1);
    acc2 = WMMA_BF16(af, f2, acc2);
    acc3 = WMMA_BF16(af, f3, acc3);
    af = afn; f0 = f0n; f1 = f1n; f2 = f2n; f3 = f3n;
  }

  const int col   = lane & 15;
  const int rbase = (lane >> 4) << 3;
  v8f accs[4] = {acc0, acc1, acc2, acc3};
#pragma unroll
  for (int u = 0; u < 4; ++u) {
    const int n = ng * 64 + u * 16 + col;
    const float bias = b[n];
#pragma unroll
    for (int j = 0; j < 8; ++j) {
      const int m = mt * 16 + rbase + j;
      G[(size_t)m * GRU_H + n] = (bf16_t)(accs[u][j] + bias);
    }
  }
}

// ---------------------------------------------------------------------------
// Phase 2a: z, r gates + rh = r*h_prev.  512 waves (one 16x16 tile each),
// pure-bf16 fragments, distance-1 pipelined K loop.
// ---------------------------------------------------------------------------
__global__ void gru_step_zr(const float* __restrict__ hprev,
                            const bf16_t* __restrict__ hbf,
                            const bf16_t* __restrict__ Wzr_h,
                            const bf16_t* __restrict__ Gz_t,
                            const bf16_t* __restrict__ Gr_t,
                            float* __restrict__ zbuf,
                            bf16_t* __restrict__ rh) {
  const int lane = threadIdx.x & 31;
  const int wave = (blockIdx.x * blockDim.x + threadIdx.x) >> 5;
  const int NT   = (2 * GRU_H) / 16;  // 128
  const int mt   = wave / NT;         // 0..3
  const int nt   = wave % NT;         // 0..127

  const bf16_t* Ab = hbf   + (size_t)(mt * 16) * GRU_H;
  const bf16_t* Bb = Wzr_h + (size_t)(nt * 16) * GRU_H;

  v8f acc = {};
  v16bf af = frag_bf16(Ab, GRU_H, lane);
  v16bf bf = frag_bf16(Bb, GRU_H, lane);
#pragma unroll 4
  for (int kb = 0; kb < GRU_H; kb += 32) {
    const int kn = (kb + 32 < GRU_H) ? (kb + 32) : 0;
    __builtin_prefetch(Bb + kb + 512, 0, 1);
    const v16bf afn = frag_bf16(Ab + kn, GRU_H, lane);
    const v16bf bfn = frag_bf16(Bb + kn, GRU_H, lane);
    acc = WMMA_BF16(af, bf, acc);
    af = afn; bf = bfn;
  }

  const int col   = lane & 15;
  const int rbase = (lane >> 4) << 3;
  const int n     = nt * 16 + col;
#pragma unroll
  for (int j = 0; j < 8; ++j) {
    const int m = mt * 16 + rbase + j;
    if (n < GRU_H) {                                   // update gate z
      const float pre = acc[j] + (float)Gz_t[(size_t)m * GRU_H + n];
      zbuf[(size_t)m * GRU_H + n] = fast_sigmoid(pre);
    } else {                                           // reset gate r -> r*h
      const int n2 = n - GRU_H;
      const float pre = acc[j] + (float)Gr_t[(size_t)m * GRU_H + n2];
      const float r = fast_sigmoid(pre);
      rh[(size_t)m * GRU_H + n2] = (bf16_t)(r * hprev[(size_t)m * GRU_H + n2]);
    }
  }
}

// ---------------------------------------------------------------------------
// Phase 2b: h~ = tanh(rh @ Wt_h.T + Gt); h_new = (1-z)h + z h~.
// Writes f32 h_new to d_out[t] AND bf16 h_new to hbf (next step's A-matrix).
// ---------------------------------------------------------------------------
__global__ void gru_step_h(const float* __restrict__ hprev,
                           const bf16_t* __restrict__ rh,
                           const bf16_t* __restrict__ Wt_h,
                           const bf16_t* __restrict__ Gt_t,
                           const float* __restrict__ zbuf,
                           float* __restrict__ hout,
                           bf16_t* __restrict__ hbf_out) {
  const int lane = threadIdx.x & 31;
  const int wave = (blockIdx.x * blockDim.x + threadIdx.x) >> 5;
  const int NT   = GRU_H / 16;   // 64
  const int mt   = wave / NT;    // 0..3
  const int nt   = wave % NT;

  const bf16_t* Ab = rh   + (size_t)(mt * 16) * GRU_H;
  const bf16_t* Bb = Wt_h + (size_t)(nt * 16) * GRU_H;

  v8f acc = {};
  v16bf af = frag_bf16(Ab, GRU_H, lane);
  v16bf bf = frag_bf16(Bb, GRU_H, lane);
#pragma unroll 4
  for (int kb = 0; kb < GRU_H; kb += 32) {
    const int kn = (kb + 32 < GRU_H) ? (kb + 32) : 0;
    __builtin_prefetch(Bb + kb + 512, 0, 1);
    const v16bf afn = frag_bf16(Ab + kn, GRU_H, lane);
    const v16bf bfn = frag_bf16(Bb + kn, GRU_H, lane);
    acc = WMMA_BF16(af, bf, acc);
    af = afn; bf = bfn;
  }

  const int col   = lane & 15;
  const int rbase = (lane >> 4) << 3;
  const int n     = nt * 16 + col;
#pragma unroll
  for (int j = 0; j < 8; ++j) {
    const int m = mt * 16 + rbase + j;
    const size_t idx = (size_t)m * GRU_H + n;
    const float pre = acc[j] + (float)Gt_t[idx];
    const float ht  = tanhf(pre);
    const float z   = zbuf[idx];
    const float hp  = hprev[idx];
    const float hn  = (1.0f - z) * hp + z * ht;
    hout[idx] = hn;
    hbf_out[idx] = (bf16_t)hn;
  }
}

// ---------------------------------------------------------------------------
extern "C" void kernel_launch(void* const* d_in, const int* in_sizes, int n_in,
                              void* d_out, int out_size, void* d_ws, size_t ws_size,
                              hipStream_t stream) {
  const float* x    = (const float*)d_in[0];  // [T, B, I]
  const float* h0   = (const float*)d_in[1];  // [B, H]
  const float* Wz_w = (const float*)d_in[2];
  const float* Wz_b = (const float*)d_in[3];
  const float* Wr_w = (const float*)d_in[4];
  const float* Wr_b = (const float*)d_in[5];
  const float* Wt_w = (const float*)d_in[6];
  const float* Wt_b = (const float*)d_in[7];
  float* out = (float*)d_out;                 // [T, B, H]

  // Workspace layout (bytes)
  char* ws = (char*)d_ws;
  const size_t SZ_WZR_H = (size_t)2 * GRU_H * GRU_H * sizeof(bf16_t);     // 4 MB
  const size_t SZ_SQ    = (size_t)GRU_H * GRU_H * sizeof(bf16_t);         // 2 MB
  const size_t SZ_G     = (size_t)GRU_T * GRU_B * GRU_H * sizeof(bf16_t); // 64 MB
  const size_t SZ_X     = (size_t)GRU_T * GRU_B * GRU_I * sizeof(bf16_t); // 64 MB

  size_t off = 0;
  bf16_t* Wzr_h = (bf16_t*)(ws + off); off += SZ_WZR_H;
  bf16_t* Wt_h  = (bf16_t*)(ws + off); off += SZ_SQ;
  bf16_t* Wzx   = (bf16_t*)(ws + off); off += SZ_SQ;
  bf16_t* Wrx   = (bf16_t*)(ws + off); off += SZ_SQ;
  bf16_t* Wtx   = (bf16_t*)(ws + off); off += SZ_SQ;
  float*  zbuf  = (float*)(ws + off);  off += (size_t)GRU_B * GRU_H * sizeof(float);
  bf16_t* rh    = (bf16_t*)(ws + off); off += (size_t)GRU_B * GRU_H * sizeof(bf16_t);
  bf16_t* hbf   = (bf16_t*)(ws + off); off += (size_t)GRU_B * GRU_H * sizeof(bf16_t);
  bf16_t* xbf   = (bf16_t*)(ws + off); off += SZ_X;
  bf16_t* Gz    = (bf16_t*)(ws + off); off += SZ_G;
  bf16_t* Gr    = (bf16_t*)(ws + off); off += SZ_G;
  bf16_t* Gt    = (bf16_t*)(ws + off); off += SZ_G;

  // Phase 0: conversions
  gru_convert_weights<<<24576, 256, 0, stream>>>(Wz_w, Wr_w, Wt_w,
                                                 Wzr_h, Wt_h, Wzx, Wrx, Wtx);
  gru_convert_x<<<32768, 256, 0, stream>>>(x, xbf);
  gru_convert_h<<<256, 256, 0, stream>>>(h0, hbf);

  // Phase 1: time-parallel input projections
  gru_gatepre<<<12288, 256, 0, stream>>>(xbf, Wzx, Wrx, Wtx,
                                         Wz_b, Wr_b, Wt_b, Gz, Gr, Gt);

  // Phase 2: sequential recurrence
  const size_t stepG = (size_t)GRU_B * GRU_H;
  for (int t = 0; t < GRU_T; ++t) {
    const float* hprev = (t == 0) ? h0 : (out + (size_t)(t - 1) * stepG);
    gru_step_zr<<<64, 256, 0, stream>>>(hprev, hbf, Wzr_h,
                                        Gz + (size_t)t * stepG,
                                        Gr + (size_t)t * stepG, zbuf, rh);
    gru_step_h<<<32, 256, 0, stream>>>(hprev, rh, Wt_h,
                                       Gt + (size_t)t * stepG, zbuf,
                                       out + (size_t)t * stepG, hbf);
  }
}
